// Nat2D_11519102288436
// MI455X (gfx1250) — compile-verified, matrix operand order checked by reading, not run on hardware
//
#include <hip/hip_runtime.h>

typedef __bf16 bf16_t;
typedef __attribute__((ext_vector_type(16))) __bf16 v16bf;
typedef __attribute__((ext_vector_type(8)))  __bf16 v8bf;
typedef __attribute__((ext_vector_type(8)))  float  v8f;
typedef __attribute__((ext_vector_type(4)))  int    v4i;

__device__ __forceinline__ v16bf make_frag(v8bf lo, v8bf hi) {
    union { v16bf v; v8bf h[2]; } u;
    u.h[0] = lo; u.h[1] = hi;
    return u.v;
}

// ---------------------------------------------------------------------------
// Async global -> LDS 16-byte copy (ASYNCcnt path), with sync fallback.
// Builtin signature (from hipcc diagnostic): params are int4* in AS1 / AS3.
// ---------------------------------------------------------------------------
#if __has_builtin(__builtin_amdgcn_global_load_async_to_lds_b128)
#define HAVE_ASYNC_LDS 1
#else
#define HAVE_ASYNC_LDS 0
#endif

__device__ __forceinline__ void async_ld16(const bf16_t* g, bf16_t* l) {
#if HAVE_ASYNC_LDS
    __builtin_amdgcn_global_load_async_to_lds_b128(
        (__attribute__((address_space(1))) v4i*)(g),
        (__attribute__((address_space(3))) v4i*)(l), 0, 0);
#else
    *(v8bf*)l = *(const v8bf*)g;
#endif
}

__device__ __forceinline__ void async_wait() {
#if HAVE_ASYNC_LDS && __has_builtin(__builtin_amdgcn_s_wait_asynccnt)
    __builtin_amdgcn_s_wait_asynccnt(0);
#endif
}

// ---------------------------------------------------------------------------
// Prep kernels: fp32 -> bf16 copies / transposes (tiny, bandwidth-trivial)
// ---------------------------------------------------------------------------
__global__ void cvt_f32_bf16(const float* __restrict__ in, bf16_t* __restrict__ out,
                             size_t n) {
    size_t i = (size_t)blockIdx.x * blockDim.x + threadIdx.x;
    if (i < n) out[i] = (bf16_t)in[i];
}

// in [R, C] fp32 -> out [C, R] bf16   (builds the [N,K] B-matrix copy)
__global__ void transpose_cvt(const float* __restrict__ in, bf16_t* __restrict__ out,
                              int R, int C) {
    int i = blockIdx.x * blockDim.x + threadIdx.x;
    if (i < R * C) {
        int r = i / C, c = i % C;
        out[(size_t)c * R + r] = (bf16_t)in[i];
    }
}

// ---------------------------------------------------------------------------
// bf16 WMMA GEMM with double-buffered async-LDS staging.
//   C[M,N] = A[M,K] * B[K,N] + bias[N]
//   A  : bf16 row-major [M,K]
//   BT : bf16 row-major [N,K]  (pre-transposed B)
//   C  : fp32 row-major [M,N]
// Block = 256 threads = 8 waves (2 M x 4 N); wave tile 64x32; block tile
// 128x128; K-step 32.  LDS: 2 x (A + B) tiles padded to 40-elem rows
// (stride 20 banks -> conflict-free b128 reads).  M,N,K multiples of
// 128/128/32 (true for this problem).
// ---------------------------------------------------------------------------
__global__ __launch_bounds__(256)
void gemm_bf16_wmma(const bf16_t* __restrict__ A, const bf16_t* __restrict__ BT,
                    const float* __restrict__ bias, float* __restrict__ C,
                    int M, int N, int K) {
    constexpr int LDT = 40;                       // padded LDS row stride (elems)
    __shared__ __align__(16) bf16_t sA[2][128 * LDT];
    __shared__ __align__(16) bf16_t sB[2][128 * LDT];

    const int tid  = threadIdx.x;
    const int lane = tid & 31;
    const int wave = tid >> 5;
    const int wm   = wave >> 2;                   // 0..1
    const int wn   = wave & 3;                    // 0..3
    const int lm   = lane & 15;
    const int lh   = lane >> 4;                   // K-half selector (ISA layout)
    const int blockM = blockIdx.x * 128;
    const int blockN = blockIdx.y * 128;

    // cooperative tile load: 256 threads x 2 chunks x 16B cover 128 rows x 64B
    const int r0 = tid >> 2;                      // rows 0..63
    const int r1 = r0 + 64;                       // rows 64..127
    const int c0 = (tid & 3) * 8;                 // k-chunk within row

    v8f acc[4][2];
#pragma unroll
    for (int i = 0; i < 4; ++i)
#pragma unroll
        for (int j = 0; j < 2; ++j)
            acc[i][j] = (v8f)(0.0f);

    auto load_tile = [&](int buf, int k0) {
        async_ld16(A  + (size_t)(blockM + r0) * K + k0 + c0, &sA[buf][r0 * LDT + c0]);
        async_ld16(A  + (size_t)(blockM + r1) * K + k0 + c0, &sA[buf][r1 * LDT + c0]);
        async_ld16(BT + (size_t)(blockN + r0) * K + k0 + c0, &sB[buf][r0 * LDT + c0]);
        async_ld16(BT + (size_t)(blockN + r1) * K + k0 + c0, &sB[buf][r1 * LDT + c0]);
    };

    auto compute = [&](int buf) {
        const int ka = 8 * lh;                    // lanes 0-15: K 0..7 ; 16-31: 8..15
        const int kb = 16 + 8 * lh;               // lanes 0-15: K 16..23; 16-31: 24..31
        v16bf afrag[4], bfrag[2];
#pragma unroll
        for (int t = 0; t < 4; ++t) {
            const bf16_t* p = &sA[buf][(wm * 64 + t * 16 + lm) * LDT];
            afrag[t] = make_frag(*(const v8bf*)(p + ka), *(const v8bf*)(p + kb));
        }
#pragma unroll
        for (int t = 0; t < 2; ++t) {
            const bf16_t* p = &sB[buf][(wn * 32 + t * 16 + lm) * LDT];
            bfrag[t] = make_frag(*(const v8bf*)(p + ka), *(const v8bf*)(p + kb));
        }
#pragma unroll
        for (int i = 0; i < 4; ++i)
#pragma unroll
            for (int j = 0; j < 2; ++j)
                acc[i][j] = __builtin_amdgcn_wmma_f32_16x16x32_bf16(
                    false, afrag[i], false, bfrag[j],
                    (short)0, acc[i][j], false, false);
    };

    load_tile(0, 0);
    async_wait();
    __syncthreads();

    const int nsteps = K / 32;
    for (int s = 0; s < nsteps; ++s) {
        const int cur = s & 1;
        if (s + 1 < nsteps) load_tile(cur ^ 1, (s + 1) * 32);
        compute(cur);
        async_wait();
        __syncthreads();
    }

    // C/D layout: VGPR r -> row 8*(lane>>4)+r, col lane&15
#pragma unroll
    for (int i = 0; i < 4; ++i)
#pragma unroll
        for (int j = 0; j < 2; ++j) {
            const int col = blockN + wn * 32 + j * 16 + lm;
            const float bv = bias[col];
            float* cp = C + (size_t)(blockM + wm * 64 + i * 16 + lh * 8) * N + col;
#pragma unroll
            for (int r = 0; r < 8; ++r)
                cp[(size_t)r * N] = acc[i][j][r] + bv;
        }
}

// ---------------------------------------------------------------------------
// Neighborhood attention, one wave32 per (b, h, i, j) query.
// qkv fp32 [B*4096, 1536]; cols [0,512)=q, [512,1024)=k, [1024,1536)=v,
// channel = h*64 + d.  Lane t owns channels {2t, 2t+1} (coalesced float2).
// Online softmax over the 49 clamped-window neighbors; output written as
// bf16 A-matrix [B*4096, 512] for the projection GEMM.
// ---------------------------------------------------------------------------
__global__ __launch_bounds__(256)
void natten2d_wave(const float* __restrict__ qkv, bf16_t* __restrict__ outA,
                   int Bsz) {
    const int lane = threadIdx.x & 31;
    const int gw   = blockIdx.x * 8 + (threadIdx.x >> 5);   // global wave id
    const int j = gw & 63;
    const int i = (gw >> 6) & 63;
    const int h = (gw >> 12) & 7;
    const int b = gw >> 15;
    if (b >= Bsz) return;

    const size_t rowq = (size_t)b * 4096 + i * 64 + j;
    const int    ch   = h * 64 + 2 * lane;

    float2 q2 = *(const float2*)(qkv + rowq * 1536 + ch);
    q2.x *= 0.125f;  // head_dim^-0.5, hd = 64
    q2.y *= 0.125f;

    const int si = min(max(i - 3, 0), 57);   // clamp(i - K/2, 0, H-K)
    const int sj = min(max(j - 3, 0), 57);

    float  mrun = -1e30f, lrun = 0.0f;
    float2 acc  = {0.0f, 0.0f};

#pragma unroll
    for (int a = 0; a < 7; ++a) {
#pragma unroll
        for (int bb = 0; bb < 7; ++bb) {
            const size_t rown = (size_t)b * 4096 + (si + a) * 64 + (sj + bb);
            const float* kp = qkv + rown * 1536 + 512 + ch;
            float2 k2 = *(const float2*)kp;

            float p = q2.x * k2.x + q2.y * k2.y;
#pragma unroll
            for (int off = 16; off > 0; off >>= 1)
                p += __shfl_xor(p, off, 32);          // logit in all lanes

            const float mnew = fmaxf(mrun, p);
            const float corr = __expf(mrun - mnew);
            const float e    = __expf(p - mnew);

            const float* vp = qkv + rown * 1536 + 1024 + ch;
            float2 v2 = *(const float2*)vp;

            acc.x = acc.x * corr + e * v2.x;
            acc.y = acc.y * corr + e * v2.y;
            lrun  = lrun * corr + e;
            mrun  = mnew;
        }
    }

    const float inv = 1.0f / lrun;
    bf16_t* op = outA + rowq * 512 + ch;
    op[0] = (bf16_t)(acc.x * inv);
    op[1] = (bf16_t)(acc.y * inv);
}

// ---------------------------------------------------------------------------
// Launch
// ---------------------------------------------------------------------------
extern "C" void kernel_launch(void* const* d_in, const int* in_sizes, int n_in,
                              void* d_out, int out_size, void* d_ws, size_t ws_size,
                              hipStream_t stream) {
    const float* x      = (const float*)d_in[0];   // [B,64,64,512]
    const float* w_qkv  = (const float*)d_in[1];   // [512,1536]
    const float* b_qkv  = (const float*)d_in[2];   // [1536]
    const float* w_proj = (const float*)d_in[3];   // [512,512]
    const float* b_proj = (const float*)d_in[4];   // [512]
    float* out = (float*)d_out;

    const int C  = 512;
    const int B  = in_sizes[0] / (64 * 64 * C);
    const int M  = B * 64 * 64;                    // 8192 for B=2

    // workspace carve-up (256B aligned)
    size_t off = 0;
    auto carve = [&](size_t bytes) {
        void* p = (char*)d_ws + off;
        off += (bytes + 255) & ~(size_t)255;
        return p;
    };
    bf16_t* x_bf    = (bf16_t*)carve((size_t)M * C * sizeof(bf16_t));
    bf16_t* wqkvT   = (bf16_t*)carve((size_t)3 * C * C * sizeof(bf16_t));
    bf16_t* wprojT  = (bf16_t*)carve((size_t)C * C * sizeof(bf16_t));
    float*  qkv     = (float*)carve((size_t)M * 3 * C * sizeof(float));
    bf16_t* attnA   = (bf16_t*)carve((size_t)M * C * sizeof(bf16_t));
    (void)ws_size;

    // 1) precision staging
    {
        size_t n = (size_t)M * C;
        cvt_f32_bf16<<<(unsigned)((n + 255) / 256), 256, 0, stream>>>(x, x_bf, n);
    }
    transpose_cvt<<<(C * 3 * C + 255) / 256, 256, 0, stream>>>(w_qkv, wqkvT, C, 3 * C);
    transpose_cvt<<<(C * C + 255) / 256, 256, 0, stream>>>(w_proj, wprojT, C, C);

    // 2) QKV GEMM: [M,512] x [512,1536] -> qkv fp32
    {
        dim3 grid(M / 128, (3 * C) / 128);
        gemm_bf16_wmma<<<grid, 256, 0, stream>>>(x_bf, wqkvT, b_qkv, qkv, M, 3 * C, C);
    }

    // 3) neighborhood attention (one wave per query), emits bf16 proj A-matrix
    {
        const int waves = B * 8 * 64 * 64;
        natten2d_wave<<<waves / 8, 256, 0, stream>>>(qkv, attnA, B);
    }

    // 4) projection GEMM: [M,512] x [512,512] + b_proj -> out fp32
    {
        dim3 grid(M / 128, C / 128);
        gemm_bf16_wmma<<<grid, 256, 0, stream>>>(attnA, wprojT, b_proj, out, M, C, C);
    }
}
